// Qwen3TTSTokenizerSingleCodebookMultiHeadAttention_12524124636036
// MI455X (gfx1250) — matrix operand order repaired and checked by reading, not compiled, not run on hardware
//
#include <hip/hip_runtime.h>
#include <hip/hip_bf16.h>
#include <stdint.h>

// ---------------------------------------------------------------------------
// MI455X (gfx1250) varlen MHA: fp32 -> bf16 WMMA (f32 accum) pipeline.
//   phase 0: convert x, Wq, Wk, Wv, Wo to bf16 in workspace
//   phase 1: fused QKV GEMM, ping-pong software pipeline, 32x64 tile / wave
//            (Q,K row-major bf16; V stored TRANSPOSED [1024,T])
//   phase 2: flash-style attention, S^T = K*Q^T trick => score accumulator
//            layout == P A-fragment layout (no cross-lane shuffles for P);
//            V loads issued before softmax to overlap v_exp_f32;
//            global_prefetch of next key block.
//   phase 3: output projection (same pipelined GEMM) -> fp32 d_out
// ---------------------------------------------------------------------------

typedef __attribute__((ext_vector_type(16))) __bf16 v16bf;
typedef __attribute__((ext_vector_type(8)))  float  v8f;

#define LOG2E 1.4426950408889634f

static __device__ __forceinline__ __bf16 f2bf(float f) {
  unsigned u = __builtin_bit_cast(unsigned, f);
  u += 0x7FFFu + ((u >> 16) & 1u);                 // round-to-nearest-even
  unsigned short s = (unsigned short)(u >> 16);
  return __builtin_bit_cast(__bf16, s);
}
static __device__ __forceinline__ unsigned pk2(float a, float b) {
  unsigned ua = __builtin_bit_cast(unsigned, a);
  ua += 0x7FFFu + ((ua >> 16) & 1u);
  unsigned ub = __builtin_bit_cast(unsigned, b);
  ub += 0x7FFFu + ((ub >> 16) & 1u);
  return (ua >> 16) | (ub & 0xFFFF0000u);
}

union Frag {                      // 16 bf16 = one WMMA A/B operand (8 VGPRs)
  struct { uint4 lo, hi; } u;
  v16bf v;
};

static __device__ __forceinline__ uint4 ld16(const __bf16* p) {
  return *(const uint4*)(p);     // global_load_b128
}

static __device__ __forceinline__ v8f wmma_bf16(v16bf a, v16bf b, v8f c) {
  return __builtin_amdgcn_wmma_f32_16x16x32_bf16(false, a, false, b,
                                                 (short)0, c, false, false);
}

// ---------------------------------------------------------------------------
// phase 0: fp32 -> bf16, 8 elements / thread
// ---------------------------------------------------------------------------
__global__ __launch_bounds__(256) void k_cvt(const float* __restrict__ s,
                                             __bf16* __restrict__ d, int n8) {
  int i = blockIdx.x * 256 + threadIdx.x;
  if (i >= n8) return;
  const float4* sp = (const float4*)s + (size_t)i * 2;
  float4 f0 = sp[0], f1 = sp[1];
  uint4 o;
  o.x = pk2(f0.x, f0.y); o.y = pk2(f0.z, f0.w);
  o.z = pk2(f1.x, f1.y); o.w = pk2(f1.z, f1.w);
  ((uint4*)d)[i] = o;
}

// ---------------------------------------------------------------------------
// phase 1: fused QKV projection.  grid = (48, T/256), block = 256 (8 waves).
// blockIdx.x: [0,16) -> Q, [16,32) -> K, [32,48) -> V.
// Each wave: 32x64 output tile (2 A-frags x 4 B-frags = 8 WMMA / k-step).
// Ping-pong two fragment sets across k: loads for step i+1 issue before the
// WMMAs of step i, and buffers alternate by name (no register rotation moves).
// A-fragment (16x32 bf16, ISA 7.12.2): lane l -> row l%16; K chunks at
//   [h*8,+8) and [16+h*8,+8), h = l/16  -> two b128 loads.
// B-fragment (32x16): lane l -> col l%16; K chunks [h*8,+8) and [16+h*8,+8), as A -> two b128.
// ---------------------------------------------------------------------------
__global__ __launch_bounds__(256) void k_qkv(
    const __bf16* __restrict__ xb,
    const __bf16* __restrict__ Wqb, const __bf16* __restrict__ Wkb,
    const __bf16* __restrict__ Wvb,
    const float* __restrict__ bq, const float* __restrict__ bv,
    __bf16* __restrict__ Qb, __bf16* __restrict__ Kb,
    __bf16* __restrict__ Vtb, int T)
{
  const int wv = threadIdx.x >> 5, lane = threadIdx.x & 31;
  const int h  = lane >> 4, ln = lane & 15;
  const int row0 = (blockIdx.y * 8 + wv) * 32;
  const int sel  = blockIdx.x >> 4;            // 0=Q 1=K 2=V
  const int nloc = (blockIdx.x & 15) * 64;
  const __bf16* W = (sel == 0) ? Wqb : (sel == 1) ? Wkb : Wvb;

  const __bf16* ar0 = xb + (size_t)(row0 + ln) * 1024 + h * 8;
  const __bf16* ar1 = ar0 + (size_t)16 * 1024;
  const __bf16* wr  = W + (size_t)(nloc + ln) * 1024 + h * 8;

  v8f acc[2][4] = {};
  Frag aA0, aA1, bA[4], aB0, aB1, bB[4];

  auto loadAB = [&](Frag& f0, Frag& f1, Frag (&bb)[4], int off) {
    f0.u.lo = ld16(ar0 + off); f0.u.hi = ld16(ar0 + off + 16);
    f1.u.lo = ld16(ar1 + off); f1.u.hi = ld16(ar1 + off + 16);
#pragma unroll
    for (int j = 0; j < 4; ++j) {
      bb[j].u.lo = ld16(wr + j * 16384 + off);
      bb[j].u.hi = ld16(wr + j * 16384 + off + 16);
    }
  };
  auto mulAB = [&](const Frag& f0, const Frag& f1, const Frag (&bb)[4]) {
#pragma unroll
    for (int j = 0; j < 4; ++j) {
      acc[0][j] = wmma_bf16(f0.v, bb[j].v, acc[0][j]);
      acc[1][j] = wmma_bf16(f1.v, bb[j].v, acc[1][j]);
    }
  };

  loadAB(aA0, aA1, bA, 0);
  for (int kk = 0; kk < 960; kk += 64) {
    loadAB(aB0, aB1, bB, kk + 32);
    mulAB(aA0, aA1, bA);
    loadAB(aA0, aA1, bA, kk + 64);
    mulAB(aB0, aB1, bB);
  }
  loadAB(aB0, aB1, bB, 992);
  mulAB(aA0, aA1, bA);        // k = 960
  mulAB(aB0, aB1, bB);        // k = 992

  if (sel == 2) {                              // V: bias + transposed store
#pragma unroll
    for (int mi = 0; mi < 2; ++mi) {
#pragma unroll
      for (int j = 0; j < 4; ++j) {
        int d = nloc + j * 16 + ln;
        float bb = bv[d];
        uint4 p;
        p.x = pk2(acc[mi][j][0] + bb, acc[mi][j][1] + bb);
        p.y = pk2(acc[mi][j][2] + bb, acc[mi][j][3] + bb);
        p.z = pk2(acc[mi][j][4] + bb, acc[mi][j][5] + bb);
        p.w = pk2(acc[mi][j][6] + bb, acc[mi][j][7] + bb);
        *(uint4*)(Vtb + (size_t)d * T + row0 + mi * 16 + h * 8) = p;
      }
    }
  } else {                                     // Q (bias) / K (no bias)
    __bf16* out = (sel == 0) ? Qb : Kb;
#pragma unroll
    for (int mi = 0; mi < 2; ++mi) {
#pragma unroll
      for (int j = 0; j < 4; ++j) {
        int col = nloc + j * 16 + ln;
        float bb = (sel == 0) ? bq[col] : 0.0f;
#pragma unroll
        for (int v = 0; v < 8; ++v)
          out[(size_t)(row0 + mi * 16 + h * 8 + v) * 1024 + col] =
              f2bf(acc[mi][j][v] + bb);
      }
    }
  }
}

// ---------------------------------------------------------------------------
// phase 2: attention.  grid = (B, 16 heads, ceil(T/128)), block = 256.
// Each wave: one 16-query tile, streams keys in 32-key steps.
// S^T = K*Q^T: C-layout (lane -> query col, VGPR -> key row) matches the
// P A-fragment layout lane-for-lane -> pack p to bf16 in-lane, no LDS.
// Assumes seqlens are multiples of 32 (true here: multiples of 128).
// ---------------------------------------------------------------------------
__global__ __launch_bounds__(256) void k_attn(
    const __bf16* __restrict__ Qb, const __bf16* __restrict__ Kb,
    const __bf16* __restrict__ Vtb, const int* __restrict__ cu,
    __bf16* __restrict__ Ctx, int T)
{
  const int seq = blockIdx.x, head = blockIdx.y;
  const int s0 = cu[seq], s1 = cu[seq + 1];
  const int L = s1 - s0;
  const int q0 = blockIdx.z * 128;
  if (q0 >= L) return;                          // uniform per block

  const int wv = threadIdx.x >> 5, lane = threadIdx.x & 31;
  const int h  = lane >> 4, ln = lane & 15;
  const int qrow = s0 + q0 + wv * 16;

  // Q^T B-fragments (hoisted): lane -> query col, channels [f*32+h*8,+8) and [f*32+16+h*8,+8)
  const __bf16* qp = Qb + (size_t)(qrow + ln) * 1024 + head * 64;
  Frag qf0, qf1;
  qf0.u.lo = ld16(qp + h * 8);        qf0.u.hi = ld16(qp + 16 + h * 8);
  qf1.u.lo = ld16(qp + 32 + h * 8);   qf1.u.hi = ld16(qp + 48 + h * 8);

  v8f o[4] = {};
  float mrun = -3.0e38f, lsum = 0.0f;
  const __bf16* kbase = Kb + (size_t)s0 * 1024 + head * 64;
  const __bf16* vbase = Vtb + s0;

  for (int kb = 0; kb < L; kb += 32) {
    const __bf16* kr0 = kbase + (size_t)(kb + ln) * 1024;
    const __bf16* kr1 = kr0 + (size_t)16 * 1024;

    // prefetch next key block (global_prefetch_b8, per-lane cacheline pull)
    if (kb + 32 < L) {
      __builtin_prefetch(kr0 + 32768, 0, 3);
      __builtin_prefetch(kr1 + 32768, 0, 3);
#pragma unroll
      for (int j = 0; j < 4; ++j)
        __builtin_prefetch(vbase + (size_t)(head * 64 + j * 16 + ln) * T
                           + kb + 32, 0, 3);
    }

    // --- S^T tiles: keys [kb,kb+16) -> sa, [kb+16,kb+32) -> sb -------------
    v8f sa = {}, sb = {};
    {
      Frag ka;
      ka.u.lo = ld16(kr0 + h * 8);      ka.u.hi = ld16(kr0 + 16 + h * 8);
      sa = wmma_bf16(ka.v, qf0.v, sa);
      ka.u.lo = ld16(kr0 + 32 + h * 8); ka.u.hi = ld16(kr0 + 48 + h * 8);
      sa = wmma_bf16(ka.v, qf1.v, sa);
      ka.u.lo = ld16(kr1 + h * 8);      ka.u.hi = ld16(kr1 + 16 + h * 8);
      sb = wmma_bf16(ka.v, qf0.v, sb);
      ka.u.lo = ld16(kr1 + 32 + h * 8); ka.u.hi = ld16(kr1 + 48 + h * 8);
      sb = wmma_bf16(ka.v, qf1.v, sb);
    }

    // --- issue V fragment loads now; they overlap the softmax VALU below ---
    Frag vf[4];
#pragma unroll
    for (int j = 0; j < 4; ++j) {
      const __bf16* vp = vbase + (size_t)(head * 64 + j * 16 + ln) * T
                         + kb + h * 8;
      vf[j].u.lo = ld16(vp); vf[j].u.hi = ld16(vp + 16);
    }

    // --- online softmax (scale 1/sqrt(64) = 0.125; +1.0 mask is invariant) -
    float tmax = -3.0e38f;
#pragma unroll
    for (int v = 0; v < 8; ++v) {
      sa[v] *= 0.125f; sb[v] *= 0.125f;
      tmax = fmaxf(tmax, fmaxf(sa[v], sb[v]));
    }
    tmax = fmaxf(tmax, __shfl_xor(tmax, 16, 32));
    float mnew = fmaxf(mrun, tmax);
    float corr = exp2f((mrun - mnew) * LOG2E);
    float psum = 0.0f;
    v16bf pf;                                   // P A-fragment, in-lane pack
#pragma unroll
    for (int v = 0; v < 8; ++v) {
      float pa = exp2f((sa[v] - mnew) * LOG2E);
      float pb = exp2f((sb[v] - mnew) * LOG2E);
      psum += pa + pb;
      pf[v]     = f2bf(pa);
      pf[8 + v] = f2bf(pb);
    }
    psum += __shfl_xor(psum, 16, 32);
    lsum = lsum * corr + psum;
    mrun = mnew;
    // --- renormalize O (O row v on this lane = query h*8+v) ----------------
#pragma unroll
    for (int v = 0; v < 8; ++v) {
      float fv = __shfl(corr, (h << 3) + v, 32);
      o[0][v] *= fv; o[1][v] *= fv; o[2][v] *= fv; o[3][v] *= fv;
    }
    // --- O += P * V --------------------------------------------------------
#pragma unroll
    for (int j = 0; j < 4; ++j)
      o[j] = wmma_bf16(pf, vf[j].v, o[j]);
  }

  float rinv = 1.0f / lsum;
#pragma unroll
  for (int v = 0; v < 8; ++v) {
    float rv = __shfl(rinv, (h << 3) + v, 32);
    int t = qrow + h * 8 + v;
#pragma unroll
    for (int j = 0; j < 4; ++j)
      Ctx[(size_t)t * 1024 + head * 64 + j * 16 + ln] = f2bf(o[j][v] * rv);
  }
}

// ---------------------------------------------------------------------------
// phase 3: out = ctx @ Wo^T + bo (fp32 output).  grid = (16, T/256).
// Same ping-pong pipelined 32x64-per-wave GEMM as phase 1.
// ---------------------------------------------------------------------------
__global__ __launch_bounds__(256) void k_oproj(
    const __bf16* __restrict__ Ctx, const __bf16* __restrict__ Wob,
    const float* __restrict__ bo, float* __restrict__ out, int T)
{
  const int wv = threadIdx.x >> 5, lane = threadIdx.x & 31;
  const int h  = lane >> 4, ln = lane & 15;
  const int row0 = (blockIdx.y * 8 + wv) * 32;
  const int nloc = blockIdx.x * 64;

  const __bf16* ar0 = Ctx + (size_t)(row0 + ln) * 1024 + h * 8;
  const __bf16* ar1 = ar0 + (size_t)16 * 1024;
  const __bf16* wr  = Wob + (size_t)(nloc + ln) * 1024 + h * 8;

  v8f acc[2][4] = {};
  Frag aA0, aA1, bA[4], aB0, aB1, bB[4];

  auto loadAB = [&](Frag& f0, Frag& f1, Frag (&bb)[4], int off) {
    f0.u.lo = ld16(ar0 + off); f0.u.hi = ld16(ar0 + off + 16);
    f1.u.lo = ld16(ar1 + off); f1.u.hi = ld16(ar1 + off + 16);
#pragma unroll
    for (int j = 0; j < 4; ++j) {
      bb[j].u.lo = ld16(wr + j * 16384 + off);
      bb[j].u.hi = ld16(wr + j * 16384 + off + 16);
    }
  };
  auto mulAB = [&](const Frag& f0, const Frag& f1, const Frag (&bb)[4]) {
#pragma unroll
    for (int j = 0; j < 4; ++j) {
      acc[0][j] = wmma_bf16(f0.v, bb[j].v, acc[0][j]);
      acc[1][j] = wmma_bf16(f1.v, bb[j].v, acc[1][j]);
    }
  };

  loadAB(aA0, aA1, bA, 0);
  for (int kk = 0; kk < 960; kk += 64) {
    loadAB(aB0, aB1, bB, kk + 32);
    mulAB(aA0, aA1, bA);
    loadAB(aA0, aA1, bA, kk + 64);
    mulAB(aB0, aB1, bB);
  }
  loadAB(aB0, aB1, bB, 992);
  mulAB(aA0, aA1, bA);        // k = 960
  mulAB(aB0, aB1, bB);        // k = 992

#pragma unroll
  for (int mi = 0; mi < 2; ++mi) {
#pragma unroll
    for (int j = 0; j < 4; ++j) {
      int col = nloc + j * 16 + ln;
      float bb = bo[col];
#pragma unroll
      for (int v = 0; v < 8; ++v)
        out[(size_t)(row0 + mi * 16 + h * 8 + v) * 1024 + col] =
            acc[mi][j][v] + bb;
    }
  }
}

// ---------------------------------------------------------------------------
extern "C" void kernel_launch(void* const* d_in, const int* in_sizes, int n_in,
                              void* d_out, int out_size, void* d_ws,
                              size_t ws_size, hipStream_t stream) {
  const float* x  = (const float*)d_in[0];
  const int*   cu = (const int*)d_in[1];
  const float* Wq = (const float*)d_in[2];
  const float* bq = (const float*)d_in[3];
  const float* Wk = (const float*)d_in[4];
  const float* Wv = (const float*)d_in[5];
  const float* bv = (const float*)d_in[6];
  const float* Wo = (const float*)d_in[7];
  const float* bo = (const float*)d_in[8];
  float* out = (float*)d_out;

  const int T = in_sizes[0] / 1024;            // 6144
  const int B = in_sizes[1] - 1;               // 8 sequences
  const size_t T1024 = (size_t)T * 1024;
  const size_t WSZ = 1024u * 1024u;

  // workspace layout (bf16), ~71 MB total
  __bf16* xb  = (__bf16*)d_ws;
  __bf16* Wqb = xb + T1024;
  __bf16* Wkb = Wqb + WSZ;
  __bf16* Wvb = Wkb + WSZ;
  __bf16* Wob = Wvb + WSZ;
  __bf16* Qb  = Wob + WSZ;
  __bf16* Kb  = Qb + T1024;
  __bf16* Vtb = Kb + T1024;                    // transposed [1024][T]
  __bf16* Ctx = Vtb + T1024;

  // phase 0: conversions
  {
    int n8 = (int)(T1024 / 8);
    k_cvt<<<(n8 + 255) / 256, 256, 0, stream>>>(x, xb, n8);
    int w8 = (int)(WSZ / 8);
    dim3 g((w8 + 255) / 256);
    k_cvt<<<g, 256, 0, stream>>>(Wq, Wqb, w8);
    k_cvt<<<g, 256, 0, stream>>>(Wk, Wkb, w8);
    k_cvt<<<g, 256, 0, stream>>>(Wv, Wvb, w8);
    k_cvt<<<g, 256, 0, stream>>>(Wo, Wob, w8);
  }
  // phase 1: QKV  (3*1024 output cols / 64 per wave-column = 48)
  k_qkv<<<dim3(48, T / 256), 256, 0, stream>>>(xb, Wqb, Wkb, Wvb, bq, bv,
                                               Qb, Kb, Vtb, T);
  // phase 2: attention (blocks past a sequence's length exit immediately)
  k_attn<<<dim3(B, 16, (T + 127) / 128), 256, 0, stream>>>(Qb, Kb, Vtb, cu,
                                                           Ctx, T);
  // phase 3: output projection
  k_oproj<<<dim3(16, T / 256), 256, 0, stream>>>(Ctx, Wob, bo, out, T);
}
